// HomeostaticPredictiveMemory_369367187859
// MI455X (gfx1250) — matrix operand order repaired and checked by reading, not compiled, not run on hardware
//
#include <hip/hip_runtime.h>
#include <hip/hip_bf16.h>
#include <math.h>

#define BB 8
#define TT 2048
#define DD 512
#define SS 4
#define GH 64

typedef __bf16 bf16_t;
typedef __attribute__((ext_vector_type(16))) __bf16 v16bf;
typedef __attribute__((ext_vector_type(8)))  float  v8f;
typedef int v4i32 __attribute__((ext_vector_type(4)));
typedef __attribute__((address_space(1))) v4i32* as1_v4i;
typedef __attribute__((address_space(3))) v4i32* as3_v4i;

union FragBF { v16bf v; uint4 q[2]; };

#if __has_builtin(__builtin_amdgcn_global_load_async_to_lds_b128)
#define HAVE_ASYNC_LDS 1
#else
#define HAVE_ASYNC_LDS 0
#endif

// 16B global -> LDS copy (async on CDNA5, sync fallback otherwise)
__device__ __forceinline__ void cp_b128(const bf16_t* g, bf16_t* l) {
#if HAVE_ASYNC_LDS
  __builtin_amdgcn_global_load_async_to_lds_b128(
      (as1_v4i)(void*)g, (as3_v4i)(void*)l, 0, 0);
#else
  *(uint4*)l = *(const uint4*)g;
#endif
}
__device__ __forceinline__ void cp_wait_all() {
#if HAVE_ASYNC_LDS
#if __has_builtin(__builtin_amdgcn_s_wait_asynccnt)
  __builtin_amdgcn_s_wait_asynccnt(0);
#else
  asm volatile("s_wait_asynccnt 0" ::: "memory");
#endif
#endif
}

// ---- WMMA fragment loaders (layouts per CDNA5 ISA 7.12.2) ----
// A 16x32 bf16: lane m=l&15, half=l>>4; two contiguous 16B chunks.
__device__ __forceinline__ v16bf load_frag_a(const bf16_t* rowp, int k0, int half) {
  FragBF f;
  f.q[0] = *(const uint4*)(rowp + k0 + 8 * half);
  f.q[1] = *(const uint4*)(rowp + k0 + 16 + 8 * half);
  return f.v;
}
// B 32x16 bf16 ([n][k] storage): lane n=l&15, half=l>>4; two contiguous 16B chunks.
__device__ __forceinline__ v16bf load_frag_b(const bf16_t* colp, int k0, int half) {
  FragBF f;
  f.q[0] = *(const uint4*)(colp + k0 + 8 * half);
  f.q[1] = *(const uint4*)(colp + k0 + 16 + 8 * half);
  return f.v;
}

// ---------------- packing kernels ----------------
__global__ void pack_bf16_k(const float* __restrict__ in, bf16_t* __restrict__ out, int n) {
  int i = blockIdx.x * 256 + threadIdx.x;
  if (i < n) out[i] = (bf16_t)in[i];
}

// in: [s][d][o] (S x D x D), out: [s][o][d]
__global__ void pack_wT_k(const float* __restrict__ in, bf16_t* __restrict__ out, int n) {
  int i = blockIdx.x * 256 + threadIdx.x;
  if (i >= n) return;
  int d = i & (DD - 1);
  int o = (i >> 9) & (DD - 1);
  int s = i >> 18;
  out[i] = (bf16_t)in[(size_t)s * DD * DD + (size_t)d * DD + o];
}

// gate_W1: [s][521][64] -> out [s][64][512] (only d<512 rows)
__global__ void pack_w1T_k(const float* __restrict__ in, bf16_t* __restrict__ out, int n) {
  int i = blockIdx.x * 256 + threadIdx.x;
  if (i >= n) return;
  int d = i & (DD - 1);
  int h = (i >> 9) & (GH - 1);
  int s = i >> 15;
  out[i] = (bf16_t)in[(size_t)s * 521 * GH + (size_t)d * GH + h];
}

// ---------------- w_enc GEMM with async-LDS double-buffered B ----------------
// grid (1024, 4), block 256 (8 waves): block = 16-row M tile x all 512 N of slot s.
// B staged per K-chunk of 32: 512 rows x 32 K, padded row stride 40 (conflict-free b128).
#define WROW 40
__global__ void __launch_bounds__(256) wenc_gemm_k(const bf16_t* __restrict__ hbf,
                                                   const bf16_t* __restrict__ wT,
                                                   const float* __restrict__ bias,
                                                   float* __restrict__ out) {
  __shared__ bf16_t bsh[2][512 * WROW];
  int tid  = threadIdx.x;
  int lane = tid & 31;
  int wq   = tid >> 5;             // N quad (64 cols)
  int mt   = blockIdx.x;
  int s    = blockIdx.y;
  int m15  = lane & 15, half = lane >> 4;
  const bf16_t* arow  = hbf + (size_t)(mt * 16 + m15) * DD;
  const bf16_t* wbase = wT + (size_t)s * DD * DD;   // [n][k]

  auto stage = [&](int c, int buf) {  // thread copies rows 2*tid, 2*tid+1 (64B each)
    int k0 = c * 32;
    const bf16_t* g0 = wbase + (size_t)(tid * 2) * DD + k0;
    bf16_t* l0 = &bsh[buf][(tid * 2) * WROW];
    cp_b128(g0,      l0);      cp_b128(g0 + 8,  l0 + 8);
    cp_b128(g0 + 16, l0 + 16); cp_b128(g0 + 24, l0 + 24);
    const bf16_t* g1 = g0 + DD;
    bf16_t* l1 = l0 + WROW;
    cp_b128(g1,      l1);      cp_b128(g1 + 8,  l1 + 8);
    cp_b128(g1 + 16, l1 + 16); cp_b128(g1 + 24, l1 + 24);
  };

  v8f acc[4] = {};
  stage(0, 0);
  cp_wait_all();
  __syncthreads();
  for (int c = 0; c < 16; ++c) {
    int buf = c & 1;
    if (c + 1 < 16) stage(c + 1, buf ^ 1);  // overlap next chunk with WMMAs
    v16bf a = load_frag_a(arow, c * 32, half);
    const bf16_t* brow0 = &bsh[buf][(wq * 64 + m15) * WROW];
#pragma unroll
    for (int q = 0; q < 4; ++q) {
      v16bf b = load_frag_b(brow0 + q * 16 * WROW, 0, half);
      acc[q] = __builtin_amdgcn_wmma_f32_16x16x32_bf16(false, a, false, b, (short)0,
                                                       acc[q], false, false);
    }
    cp_wait_all();
    __syncthreads();
  }
#pragma unroll
  for (int q = 0; q < 4; ++q) {
    int n = wq * 64 + q * 16 + m15;
    float bv = bias[s * DD + n];
#pragma unroll
    for (int j = 0; j < 8; ++j) {
      int r = mt * 16 + j + 8 * half;
      out[(((size_t)r * SS + s) << 9) + n] = acc[q][j] + bv;
    }
  }
}

// ---------------- gate pre-hidden GEMM: out[b][t][s][h] = h @ W1[:D] + b1 ----------------
// grid (128, 4), block 256: 8 waves, wave = M tile
__global__ void __launch_bounds__(256) gatepre_gemm_k(const bf16_t* __restrict__ hbf,
                                                      const bf16_t* __restrict__ w1t,
                                                      const float* __restrict__ b1,
                                                      float* __restrict__ out) {
  int lane = threadIdx.x & 31;
  int wid  = threadIdx.x >> 5;
  int mt   = blockIdx.x * 8 + wid;
  int s    = blockIdx.y;
  int m15  = lane & 15, half = lane >> 4;
  const bf16_t* arow  = hbf + (size_t)(mt * 16 + m15) * DD;
  const bf16_t* brow0 = w1t + ((size_t)s * GH + m15) * DD;
  v8f acc[4] = {};
  for (int k0 = 0; k0 < DD; k0 += 32) {
    v16bf a = load_frag_a(arow, k0, half);
#pragma unroll
    for (int q = 0; q < 4; ++q) {
      v16bf b = load_frag_b(brow0 + (size_t)q * 16 * DD, k0, half);
      acc[q] = __builtin_amdgcn_wmma_f32_16x16x32_bf16(false, a, false, b, (short)0,
                                                       acc[q], false, false);
    }
  }
#pragma unroll
  for (int q = 0; q < 4; ++q) {
    int n = q * 16 + m15;
    float bv = b1[s * GH + n];
#pragma unroll
    for (int j = 0; j < 8; ++j) {
      int r = mt * 16 + j + 8 * half;
      out[((size_t)r * SS + s) * GH + n] = acc[q][j] + bv;
    }
  }
}

// ---------------- fused pred GEMM + surprise reduction, async-LDS B ----------------
// err[t+1][b][s] = || h[b,t] @ pred_W[s] + pred_b[s] - h[b,t+1] ||
// grid (128, 4), block 256: wave = M tile; block cooperatively stages each 16-col
// B slab (16 rows x 512 K, padded stride 520 -> conflict-free b128 reads).
#define EROW 520
__global__ void __launch_bounds__(256) err_gemm_k(const bf16_t* __restrict__ hbf,
                                                  const bf16_t* __restrict__ pWt,
                                                  const float* __restrict__ pb,
                                                  const float* __restrict__ hseq,
                                                  float* __restrict__ err) {
  __shared__ bf16_t bsh[2][16 * EROW];
  int tid  = threadIdx.x;
  int lane = tid & 31;
  int wid  = tid >> 5;
  int mt   = blockIdx.x * 8 + wid;
  int s    = blockIdx.y;
  int m15  = lane & 15, half = lane >> 4;
  const bf16_t* arow  = hbf + (size_t)(mt * 16 + m15) * DD;
  const bf16_t* wbase = pWt + (size_t)s * DD * DD;

  auto stage = [&](int nt, int buf) {  // thread: row tid>>4, 64B part tid&15
    int r16 = tid >> 4, part = tid & 15;
    const bf16_t* g = wbase + (size_t)(nt * 16 + r16) * DD + part * 32;
    bf16_t* l = &bsh[buf][r16 * EROW + part * 32];
    cp_b128(g,      l);      cp_b128(g + 8,  l + 8);
    cp_b128(g + 16, l + 16); cp_b128(g + 24, l + 24);
  };

  float psum[8] = {0.f, 0.f, 0.f, 0.f, 0.f, 0.f, 0.f, 0.f};
  stage(0, 0);
  cp_wait_all();
  __syncthreads();
  for (int nt = 0; nt < 32; ++nt) {
    int buf = nt & 1;
    if (nt + 1 < 32) stage(nt + 1, buf ^ 1);
    const bf16_t* brow = &bsh[buf][m15 * EROW];
    v8f acc = {};
    for (int k0 = 0; k0 < DD; k0 += 32) {
      v16bf a = load_frag_a(arow, k0, half);
      v16bf b = load_frag_b(brow, k0, half);
      acc = __builtin_amdgcn_wmma_f32_16x16x32_bf16(false, a, false, b, (short)0,
                                                    acc, false, false);
    }
    int col = nt * 16 + m15;
    float bv = pb[s * DD + col];
#pragma unroll
    for (int j = 0; j < 8; ++j) {
      int r = mt * 16 + j + 8 * half;
      int t = r & (TT - 1);
      if (t < TT - 1) {
        float tgt = hseq[((size_t)r + 1) * DD + col];
        float d = acc[j] + bv - tgt;
        psum[j] += d * d;
      }
    }
    cp_wait_all();
    __syncthreads();
  }
#pragma unroll
  for (int j = 0; j < 8; ++j) {
    float p = psum[j];
    p += __shfl_xor(p, 1, 32);
    p += __shfl_xor(p, 2, 32);
    p += __shfl_xor(p, 4, 32);
    p += __shfl_xor(p, 8, 32);
    psum[j] = p;
  }
  if (m15 == 0) {
#pragma unroll
    for (int j = 0; j < 8; ++j) {
      int r = mt * 16 + j + 8 * half;
      int t = r & (TT - 1);
      int b = r >> 11;
      if (t < TT - 1)
        err[(size_t)(t + 1) * (BB * SS) + b * SS + s] = sqrtf(psum[j] + 1e-8f);
    }
  }
}

// ---------------- sequential gate / homeostasis chain ----------------
// <<<1,128>>>: wave = slot s; lane: b = l>>2 (8 batches), hq = l&3 (16 hidden dims each)
__global__ void __launch_bounds__(128) gate_seq_k(const float* __restrict__ gatepre,
                                                  const float* __restrict__ errb,
                                                  const float* __restrict__ gW1,
                                                  const float* __restrict__ gW2,
                                                  const float* __restrict__ gb2,
                                                  const float* __restrict__ semb,
                                                  float* __restrict__ geff) {
  int lane = threadIdx.x & 31;
  int s    = threadIdx.x >> 5;
  if (s >= SS) return;
  int b  = lane >> 2;
  int hq = lane & 3;
  int h0 = hq * 16;
  const float* W1s = gW1 + (size_t)s * 521 * GH;
  float w1z[16], W2c[16], se0[16], se1[16], se2[16];
#pragma unroll
  for (int i = 0; i < 16; ++i) {
    int h = h0 + i;
    w1z[i] = W1s[512 * GH + h];
    W2c[i] = gW2[s * GH + h];
    float a0 = 0.f, a1 = 0.f, a2 = 0.f;
    for (int e = 0; e < 8; ++e) {
      float w = W1s[(513 + e) * GH + h];
      a0 += semb[e] * w; a1 += semb[8 + e] * w; a2 += semb[16 + e] * w;
    }
    se0[i] = a0; se1[i] = a1; se2[i] = a2;
  }
  float b2 = gb2[s];
  float mu = 0.f, sigma = 1.f, ge = 0.5f;
  int st = 0;

  const float* gp0 = gatepre + ((size_t)b * TT) * (SS * GH) + s * GH + h0;
  float4 gp[4];
  { const float4* p = (const float4*)gp0;
    gp[0] = p[0]; gp[1] = p[1]; gp[2] = p[2]; gp[3] = p[3]; }

  for (int t = 0; t < TT; ++t) {
    float4 gn[4];
    if (t + 1 < TT) {  // prefetch next step while this step's serial chain runs
      const float4* p = (const float4*)(gp0 + (size_t)(t + 1) * (SS * GH));
      gn[0] = p[0]; gn[1] = p[1]; gn[2] = p[2]; gn[3] = p[3];
    }
    float e = (t == 0) ? 0.f : errb[(size_t)t * (BB * SS) + b * SS + s];
    float z = (e - mu) / fmaxf(sigma, 1e-3f);
    const float* sep = (st == 0) ? se0 : (st == 1) ? se1 : se2;
    const float* gpf = (const float*)gp;
    float acc = 0.f;
#pragma unroll
    for (int i = 0; i < 16; ++i) {
      float hid = gpf[i] + z * w1z[i] + sep[i];
      hid = fmaxf(hid, 0.f);
      acc += hid * W2c[i];
    }
    acc += __shfl_xor(acc, 1, 32);
    acc += __shfl_xor(acc, 2, 32);          // full 64-dim dot per (b,s)
    float g = 1.f / (1.f + __expf(-(acc + b2)));
    float gain = (st == 0) ? 1.0f : (st == 1) ? 0.5f : 0.1f;
    if (hq == 0) geff[((size_t)b * SS + s) * TT + t] = g * gain;
    // batch means across b (lanes differing in bits 2..4)
    float gm = g, em = e, am = fabsf(e - mu);
#pragma unroll
    for (int off = 4; off < 32; off <<= 1) {
      gm += __shfl_xor(gm, off, 32);
      em += __shfl_xor(em, off, 32);
      am += __shfl_xor(am, off, 32);
    }
    gm *= 0.125f; em *= 0.125f; am *= 0.125f;
    if (t > 0) { mu = 0.99f * mu + 0.01f * em; sigma = 0.99f * sigma + 0.01f * am; }
    ge = 0.99f * ge + 0.01f * gm;
    int ns = st;
    if (st == 0 && ge < 0.1f) ns = 1;
    else if (st == 1) { if (ge < 0.03f) ns = 2; else if (ge > 0.25f) ns = 0; }
    else if (st == 2 && ge > 0.25f) ns = 0;
    st = ns;
    gp[0] = gn[0]; gp[1] = gn[1]; gp[2] = gn[2]; gp[3] = gn[3];
  }
}

// ---------------- parallel memory recurrence: 16384 independent chains ----------------
// grid 32 (b,s), block 128: thread owns 4 d's -> b128 load/store per step
__global__ void __launch_bounds__(128) scan_mem_k(const float* __restrict__ wenc,
                                                  const float* __restrict__ geff,
                                                  const float* __restrict__ w0,
                                                  float* __restrict__ out) {
  int b = blockIdx.x >> 2, s = blockIdx.x & 3;
  int d4 = threadIdx.x;  // 0..127
  float4 m = *(const float4*)(w0 + s * DD + d4 * 4);
  const float* gp = geff + ((size_t)b * SS + s) * TT;
  const float4* wp = (const float4*)(wenc + ((size_t)b * TT * SS + s) * DD) + d4;
  float4* op = (float4*)(out + (size_t)b * TT * (SS * DD) + s * DD) + d4;
#pragma unroll 4
  for (int t = 0; t < TT; ++t) {
    float g = gp[t];                        // uniform -> scalar load
    float4 w = wp[(size_t)t * (SS * DD / 4)];
    m.x += g * (w.x - m.x);
    m.y += g * (w.y - m.y);
    m.z += g * (w.z - m.z);
    m.w += g * (w.w - m.w);
    op[(size_t)t * (SS * DD / 4)] = m;
  }
}

extern "C" void kernel_launch(void* const* d_in, const int* in_sizes, int n_in,
                              void* d_out, int out_size, void* d_ws, size_t ws_size,
                              hipStream_t stream) {
  const float* h_seq   = (const float*)d_in[0];
  const float* pred_W  = (const float*)d_in[1];
  const float* pred_b  = (const float*)d_in[2];
  const float* gate_W1 = (const float*)d_in[3];
  const float* gate_b1 = (const float*)d_in[4];
  const float* gate_W2 = (const float*)d_in[5];
  const float* gate_b2 = (const float*)d_in[6];
  const float* write_W = (const float*)d_in[7];
  const float* write_b = (const float*)d_in[8];
  const float* w0      = (const float*)d_in[9];
  const float* semb    = (const float*)d_in[10];
  float* out = (float*)d_out;

  char* ws = (char*)d_ws;
  size_t off = 0;
  auto take = [&](size_t bytes) -> char* {
    char* p = ws + off;
    off += (bytes + 255) & ~(size_t)255;
    return p;
  };
  bf16_t* hbf     = (bf16_t*)take((size_t)BB * TT * DD * 2);        // 16 MB
  bf16_t* predWt  = (bf16_t*)take((size_t)SS * DD * DD * 2);        // 2 MB
  bf16_t* writeWt = (bf16_t*)take((size_t)SS * DD * DD * 2);        // 2 MB
  bf16_t* w1t     = (bf16_t*)take((size_t)SS * GH * DD * 2);        // 256 KB
  float*  wenc    = (float*)take((size_t)BB * TT * SS * DD * 4);    // 128 MB
  float*  gatepre = (float*)take((size_t)BB * TT * SS * GH * 4);    // 16 MB
  float*  errb    = (float*)take((size_t)TT * BB * SS * 4);         // 256 KB
  float*  geff    = (float*)take((size_t)BB * SS * TT * 4);         // 256 KB
  (void)ws_size; (void)in_sizes; (void)n_in; (void)out_size;

  int nh = BB * TT * DD;
  pack_bf16_k<<<dim3((nh + 255) / 256), dim3(256), 0, stream>>>(h_seq, hbf, nh);
  int nw = SS * DD * DD;
  pack_wT_k<<<dim3((nw + 255) / 256), dim3(256), 0, stream>>>(pred_W, predWt, nw);
  pack_wT_k<<<dim3((nw + 255) / 256), dim3(256), 0, stream>>>(write_W, writeWt, nw);
  int n1 = SS * GH * DD;
  pack_w1T_k<<<dim3((n1 + 255) / 256), dim3(256), 0, stream>>>(gate_W1, w1t, n1);

  wenc_gemm_k<<<dim3(BB * TT / 16, SS), dim3(256), 0, stream>>>(hbf, writeWt, write_b, wenc);
  gatepre_gemm_k<<<dim3(BB * TT / 16 / 8, SS), dim3(256), 0, stream>>>(hbf, w1t, gate_b1, gatepre);
  err_gemm_k<<<dim3(BB * TT / 16 / 8, SS), dim3(256), 0, stream>>>(hbf, predWt, pred_b, h_seq, errb);

  gate_seq_k<<<dim3(1), dim3(128), 0, stream>>>(gatepre, errb, gate_W1, gate_W2, gate_b2, semb, geff);
  scan_mem_k<<<dim3(32), dim3(128), 0, stream>>>(wenc, geff, w0, out);
}
